// GaussianKernelEncoder_13898514170542
// MI455X (gfx1250) — matrix operand order repaired and checked by reading, not compiled, not run on hardware
//
#include <hip/hip_runtime.h>

typedef __attribute__((ext_vector_type(16))) _Float16     v16h;
typedef __attribute__((ext_vector_type(8)))  float        v8f;
typedef __attribute__((ext_vector_type(8)))  unsigned int v8u;

#define BM   128   // rows of x per block
#define BN   256   // centers per LDS chunk
#define DD   64    // feature dim
#define KTOT 1024  // total centers
#define SA   72    // padded LDS row stride (halves) to avoid bank conflicts

union FragH { v8u u; v16h h; };

__global__ __launch_bounds__(256)
void gauss_wmma_kernel(const float* __restrict__ x,
                       const float* __restrict__ c,
                       float* __restrict__ out)
{
    __shared__ _Float16 xt[BM * SA];   // x tile, f16
    __shared__ _Float16 ct[BN * SA];   // centers chunk, f16
    __shared__ float    xsq[BM];
    __shared__ float    csq[BN];

    const int tid  = threadIdx.x;
    const int lane = tid & 31;
    const int wave = tid >> 5;
    const int h    = lane >> 4;   // lane half (0: lanes 0-15, 1: lanes 16-31)
    const int l16  = lane & 15;
    const long rowBase = (long)blockIdx.x * BM;

    // ---- stage x tile: fp32 global -> f16 LDS (float4 loads, packed b64 stores)
    {
        const float4* xv = (const float4*)(x + rowBase * DD);
        #pragma unroll
        for (int j = 0; j < (BM * DD / 4) / 256; ++j) {
            int f   = tid + j * 256;
            float4 v = xv[f];
            int row = f >> 4;            // f / (DD/4)
            int col = (f & 15) << 2;
            union { _Float16 q[4]; unsigned int u[2]; } p;
            p.q[0] = (_Float16)v.x; p.q[1] = (_Float16)v.y;
            p.q[2] = (_Float16)v.z; p.q[3] = (_Float16)v.w;
            *(unsigned int*)&xt[row * SA + col]     = p.u[0];
            *(unsigned int*)&xt[row * SA + col + 2] = p.u[1];
        }
    }
    __syncthreads();

    // ---- per-row squared norms of x (from the same f16 data used by WMMA)
    if (tid < BM) {
        float s = 0.f;
        #pragma unroll
        for (int k = 0; k < DD; ++k) { float v = (float)xt[tid * SA + k]; s += v * v; }
        xsq[tid] = s;
    }

    // ---- A fragments (constant across all center chunks)
    // ISA 16-bit A 16x32 layout: lanes0-15 K={0..7,16..23}, lanes16-31 K={8..15,24..31}
    FragH fa0, fa1;
    {
        const unsigned int* arw =
            (const unsigned int*)&xt[(wave * 16 + l16) * SA];
        #pragma unroll
        for (int v = 0; v < 8; ++v) {
            int kw = ((v < 4) ? v : v + 4) + 4 * h;   // (kb/2)
            fa0.u[v] = arw[kw];
            fa1.u[v] = arw[16 + kw];                  // +32 halves
        }
    }

    const int nChunks = KTOT / BN;
    for (int chunk = 0; chunk < nChunks; ++chunk) {
        __syncthreads();  // protect ct/csq reuse (and xsq first use)

        // ---- stage centers chunk fp32 -> f16 LDS
        const float4* cv = (const float4*)(c + (long)chunk * BN * DD);
        #pragma unroll
        for (int j = 0; j < (BN * DD / 4) / 256; ++j) {
            int f   = tid + j * 256;
            float4 v = cv[f];
            int row = f >> 4;
            int col = (f & 15) << 2;
            union { _Float16 q[4]; unsigned int u[2]; } p;
            p.q[0] = (_Float16)v.x; p.q[1] = (_Float16)v.y;
            p.q[2] = (_Float16)v.z; p.q[3] = (_Float16)v.w;
            *(unsigned int*)&ct[row * SA + col]     = p.u[0];
            *(unsigned int*)&ct[row * SA + col + 2] = p.u[1];
        }
        if (chunk + 1 < nChunks)  // global_prefetch_b8 of next chunk
            __builtin_prefetch(c + (long)(chunk + 1) * BN * DD + tid * 64, 0, 0);
        __syncthreads();

        // ---- center squared norms for this chunk
        {
            float s = 0.f;
            #pragma unroll
            for (int k = 0; k < DD; ++k) { float v = (float)ct[tid * SA + k]; s += v * v; }
            csq[tid] = s;
        }
        __syncthreads();

        // per-lane x-norms for the 8 C rows this lane owns (M = r + 8*h)
        float xr[8];
        #pragma unroll
        for (int r = 0; r < 8; ++r) xr[r] = xsq[wave * 16 + 8 * h + r];

        // ---- 16 N-tiles of 16 centers each
        for (int nt = 0; nt < BN / 16; ++nt) {
            // B 32x16 f16 layout: lane holds column n=l16; lane-half h holds
            // K={8h..8h+7, 16+8h..16+8h+7} (same K order as A)
            FragH fb0, fb1;
            const unsigned int* brw =
                (const unsigned int*)&ct[(nt * 16 + l16) * SA];
            #pragma unroll
            for (int e = 0; e < 8; ++e) {
                int kw = ((e < 4) ? e : e + 4) + 4 * h;
                fb0.u[e] = brw[kw];
                fb1.u[e] = brw[16 + kw];   // K += 32
            }

            v8f acc = {};
            acc = __builtin_amdgcn_wmma_f32_16x16x32_f16(
                false, fa0.h, false, fb0.h, (short)0, acc, false, false);
            acc = __builtin_amdgcn_wmma_f32_16x16x32_f16(
                false, fa1.h, false, fb1.h, (short)0, acc, false, false);

            const int   n  = chunk * BN + nt * 16 + l16;
            const float cs = csq[nt * 16 + l16];
            #pragma unroll
            for (int r = 0; r < 8; ++r) {
                float sq = xr[r] + cs - 2.0f * acc[r];
                sq = fmaxf(sq, 0.0f);
                long row = rowBase + wave * 16 + 8 * h + r;
                // exp(-sq/2) = exp2(-sq * 0.5*log2(e))
                out[row * KTOT + n] =
                    __builtin_amdgcn_exp2f(sq * -0.7213475204444817f);
            }
        }
    }
}

extern "C" void kernel_launch(void* const* d_in, const int* in_sizes, int n_in,
                              void* d_out, int out_size, void* d_ws, size_t ws_size,
                              hipStream_t stream) {
    const float* x = (const float*)d_in[0];   // [65536, 64] fp32
    const float* c = (const float*)d_in[1];   // [1024, 64]  fp32
    float* out = (float*)d_out;               // [65536, 1024] fp32

    const int B = 65536;
    dim3 grid(B / BM);     // 512 blocks
    dim3 block(256);       // 8 wave32 waves
    gauss_wmma_kernel<<<grid, block, 0, stream>>>(x, c, out);
}
